// AttentionBlock_12438225289592
// MI455X (gfx1250) — matrix operand order repaired and checked by reading, not compiled, not run on hardware
//
#include <hip/hip_runtime.h>
#include <hip/hip_bf16.h>
#include <math.h>

// ---------------------------------------------------------------------------
// MI455X (gfx1250) attention block.
// Compute-bound (~34.5 GFLOP vs ~90MB working set, fits in 192MB L2):
// all GEMMs + attention through v_wmma_f32_16x16x32_f16 (wave32), with
// CDNA5 async global->LDS staging (ASYNCcnt) for the GEMM A tiles.
// ---------------------------------------------------------------------------

typedef _Float16 f16;
typedef __attribute__((ext_vector_type(16))) _Float16 v16h;
typedef __attribute__((ext_vector_type(8)))  _Float16 v8h;
typedef __attribute__((ext_vector_type(8)))  float    v8f;

union V16 { v16h v; v8h h[2]; };

#define TOTAL_Q   2048
#define TOTAL_KV  2048
#define QIN       1024
#define KVIN      1033
#define KVPAD     1056   // next multiple of 32 above 1033
#define DQK       1024
#define DV        1024
#define NHEADS    16
#define HEADD     64
#define SCALER    0.125f // 1/sqrt(1024/16)

// CDNA5 async global->LDS copy (16B per lane), tracked by ASYNCcnt.
// VDST = VGPR holding LDS byte address, VADDR = 64-bit global address.
__device__ __forceinline__ void async_ld_b128(unsigned int lds_byte_addr,
                                              const void* gaddr) {
  asm volatile("global_load_async_to_lds_b128 %0, %1, off"
               :: "v"(lds_byte_addr),
                  "v"((unsigned long long)(uintptr_t)gaddr)
               : "memory");
}
__device__ __forceinline__ void wait_asynccnt0() {
  asm volatile("s_wait_asynccnt 0x0" ::: "memory");
}

// ---------------------------------------------------------------------------
// f32 -> f16 convert with zero padding (rows and cols).
// ---------------------------------------------------------------------------
__global__ void cvt_pad_f16(const float* __restrict__ src, f16* __restrict__ dst,
                            int sR, int sC, int dR, int dC) {
  int total = dR * dC;
  for (int idx = blockIdx.x * blockDim.x + threadIdx.x; idx < total;
       idx += gridDim.x * blockDim.x) {
    int r = idx / dC, c = idx % dC;
    float v = (r < sR && c < sC) ? src[(size_t)r * sC + c] : 0.0f;
    dst[idx] = (f16)v;
  }
}

// dst[c*R + r] = src[r*C + c]
__global__ void transpose_f16(const f16* __restrict__ src, f16* __restrict__ dst,
                              int R, int C) {
  int total = R * C;
  for (int idx = blockIdx.x * blockDim.x + threadIdx.x; idx < total;
       idx += gridDim.x * blockDim.x) {
    int r = idx / C, c = idx % C;
    dst[(size_t)c * R + r] = src[idx];
  }
}

// ---------------------------------------------------------------------------
// Tiled WMMA GEMM: C[M,N] = A[M,K] * B[K,N] + bias (A,B f16 row-major).
// Block = 128 threads = 4 wave32; block tile 64x128, 32x64 per wave
// (8 v_wmma per wave per K-step of 32).
//  - A tile (64x32) staged row-major into LDS via global_load_async_to_lds_b128
//    (ASYNCcnt); A fragments then read as two contiguous b128 ds loads
//    (lane=row m, K=16*halflane+e contiguous, same order as B).
//  - B tile (32x128) staged transposed into LDS so B fragments
//    (lane=column n, K=16*halflane+e contiguous) are two b128 ds reads.
// ---------------------------------------------------------------------------
template <bool OUT_F16>
__global__ __launch_bounds__(128)
void gemm_wmma_bias(const f16* __restrict__ A, const f16* __restrict__ B,
                    const float* __restrict__ bias, void* __restrict__ Cp,
                    int K, int lda, int ldb, int ldc) {
  __shared__ f16 shA[64][32];                 // [m within tile][k], row-major
  __shared__ f16 shB[128][32];                // [n within tile][k], transposed
  const int lane = threadIdx.x & 31;
  const int wave = threadIdx.x >> 5;
  const int ln = lane & 15, lh = lane >> 4;
  const int wm = wave >> 1, wn = wave & 1;
  const int mBlock = blockIdx.y * 64;
  const int nBlock = blockIdx.x * 128;
  const int mBase  = mBlock + wm * 32;
  const int nBase  = nBlock + wn * 64;

  // A staging: 64 rows x 64B = 256 x 16B granules, 2 per thread.
  const unsigned int shA_base = (unsigned int)(uintptr_t)(&shA[0][0]);
  const int g0 = threadIdx.x * 2;
  const int rowA0 = g0 >> 2, c0 = g0 & 3;         // c = 16B chunk within row
  const int rowA1 = (g0 + 1) >> 2, c1 = (g0 + 1) & 3;

  const int tn = threadIdx.x;                      // B staging column 0..127

  v8f acc[2][4] = {};

  for (int k0 = 0; k0 < K; k0 += 32) {
    __syncthreads();
    // --- async stage A tile (row-major, no transpose needed) ---
    async_ld_b128(shA_base + (unsigned)(rowA0 * 64 + c0 * 16),
                  A + (size_t)(mBlock + rowA0) * lda + k0 + c0 * 8);
    async_ld_b128(shA_base + (unsigned)(rowA1 * 64 + c1 * 16),
                  A + (size_t)(mBlock + rowA1) * lda + k0 + c1 * 8);
    // --- stage B tile transposed (coalesced b16 loads, merged ds stores) ---
#pragma unroll
    for (int k = 0; k < 32; ++k)
      shB[tn][k] = B[(size_t)(k0 + k) * ldb + nBlock + tn];
    if (k0 + 32 < K)                               // CDNA5 global_prefetch_b8
      __builtin_prefetch(&B[(size_t)(k0 + 32) * ldb + nBlock + tn], 0, 1);
    wait_asynccnt0();
    __syncthreads();

    V16 afrag[2], bfrag[4];
#pragma unroll
    for (int mt = 0; mt < 2; ++mt) {
      const f16* ap = &shA[wm * 32 + 16 * mt + ln][0];
      afrag[mt].h[0] = *(const v8h*)(ap + 16 * lh);
      afrag[mt].h[1] = *(const v8h*)(ap + 16 * lh + 8);
    }
#pragma unroll
    for (int nt = 0; nt < 4; ++nt) {
      const f16* bp = &shB[wn * 64 + 16 * nt + ln][16 * lh];
      bfrag[nt].h[0] = *(const v8h*)(bp);
      bfrag[nt].h[1] = *(const v8h*)(bp + 8);
    }
#pragma unroll
    for (int mt = 0; mt < 2; ++mt)
#pragma unroll
      for (int nt = 0; nt < 4; ++nt)
        acc[mt][nt] = __builtin_amdgcn_wmma_f32_16x16x32_f16(
            false, afrag[mt].v, false, bfrag[nt].v, (short)0, acc[mt][nt],
            false, false);
  }

#pragma unroll
  for (int mt = 0; mt < 2; ++mt)
#pragma unroll
    for (int nt = 0; nt < 4; ++nt) {
      const int n = nBase + 16 * nt + ln;
      const float bv = bias ? bias[n] : 0.0f;
#pragma unroll
      for (int r = 0; r < 8; ++r) {
        const int row = mBase + 16 * mt + r + 8 * lh;
        const float val = acc[mt][nt][r] + bv;
        if (OUT_F16)
          ((f16*)Cp)[(size_t)row * ldc + n] = (f16)val;
        else
          ((float*)Cp)[(size_t)row * ldc + n] = val;
      }
    }
}

// ---------------------------------------------------------------------------
// Flash attention, one wave32 per (16-query tile, head).
// S tile (16q x 16k) = two chained 16x16x32 WMMAs over d=64.
// Segment mask + online softmax on C-layout registers (lane = key column,
// vgpr r = query row r + 8*(lane>>4)); P relayout C->A via 1KB LDS tile;
// P*V uses VT (pre-transposed V) so B fragments are contiguous loads.
// -1e30 mask fill is self-correcting under online softmax rescaling.
// ---------------------------------------------------------------------------
__global__ __launch_bounds__(32)
void flash_attn_wmma(const f16* __restrict__ Qh, const f16* __restrict__ Kh,
                     const f16* __restrict__ VT, const int* __restrict__ seg_q,
                     const int* __restrict__ seg_kv, f16* __restrict__ WVh) {
  __shared__ f16 lds_p[16 * 32];
  const int lane = threadIdx.x & 31;
  const int ln = lane & 15, lh = lane >> 4;
  const int q0 = blockIdx.x * 16;
  const int h  = blockIdx.y;

  // Q A-fragments for K-steps 0 and 32 of the 64-wide head dim.
  V16 qa[2];
  const f16* qrow = Qh + (size_t)(q0 + ln) * DQK + h * HEADD;
  qa[0].h[0] = *(const v8h*)(qrow + 16 * lh);
  qa[0].h[1] = *(const v8h*)(qrow + 16 * lh + 8);
  qa[1].h[0] = *(const v8h*)(qrow + 32 + 16 * lh);
  qa[1].h[1] = *(const v8h*)(qrow + 32 + 16 * lh + 8);

  int sq[8];
#pragma unroll
  for (int r = 0; r < 8; ++r) sq[r] = seg_q[q0 + 8 * lh + r];

  float m_i[8], l_i[8];
#pragma unroll
  for (int r = 0; r < 8; ++r) { m_i[r] = -1e30f; l_i[r] = 0.0f; }
  v8f acc[4] = {};  // 16 x 64 output accumulator (4 n-tiles of 16)

  for (int kb = 0; kb < TOTAL_KV; kb += 32) {
    float s[2][8];
#pragma unroll
    for (int t = 0; t < 2; ++t) {
      const int key = kb + 16 * t + ln;
      const f16* krow = Kh + (size_t)key * DQK + h * HEADD;
      V16 kf0, kf1;
      kf0.h[0] = *(const v8h*)(krow + 16 * lh);
      kf0.h[1] = *(const v8h*)(krow + 16 * lh + 8);
      kf1.h[0] = *(const v8h*)(krow + 32 + 16 * lh);
      kf1.h[1] = *(const v8h*)(krow + 32 + 16 * lh + 8);
      v8f sv = {};
      sv = __builtin_amdgcn_wmma_f32_16x16x32_f16(false, qa[0].v, false, kf0.v,
                                                  (short)0, sv, false, false);
      sv = __builtin_amdgcn_wmma_f32_16x16x32_f16(false, qa[1].v, false, kf1.v,
                                                  (short)0, sv, false, false);
      const int sk = seg_kv[key];
#pragma unroll
      for (int r = 0; r < 8; ++r) {
        const float val = sv[r] * SCALER;
        s[t][r] = (sk == sq[r]) ? val : -1e30f;
      }
    }

    // Online softmax: per-row max / sum via shfl_xor across the 16-lane group.
    float sc[8];
#pragma unroll
    for (int r = 0; r < 8; ++r) {
      float bm = fmaxf(s[0][r], s[1][r]);
#pragma unroll
      for (int o = 8; o >= 1; o >>= 1) bm = fmaxf(bm, __shfl_xor(bm, o, 32));
      const float m_new = fmaxf(m_i[r], bm);
      sc[r] = __expf(m_i[r] - m_new);
      const float p0 = __expf(s[0][r] - m_new);
      const float p1 = __expf(s[1][r] - m_new);
      s[0][r] = p0; s[1][r] = p1;
      float ps = p0 + p1;
#pragma unroll
      for (int o = 8; o >= 1; o >>= 1) ps += __shfl_xor(ps, o, 32);
      l_i[r] = l_i[r] * sc[r] + ps;
      m_i[r] = m_new;
    }

    // Relayout P (C layout -> A layout) through LDS.
    __syncthreads();
#pragma unroll
    for (int t = 0; t < 2; ++t)
#pragma unroll
      for (int r = 0; r < 8; ++r)
        lds_p[(8 * lh + r) * 32 + 16 * t + ln] = (f16)s[t][r];
    __syncthreads();
    V16 pa;
    const f16* pp = lds_p + ln * 32;
    pa.h[0] = *(const v8h*)(pp + 16 * lh);
    pa.h[1] = *(const v8h*)(pp + 16 * lh + 8);

    // Rescale accumulator, then P (16x32) x V (32x64).
#pragma unroll
    for (int nt = 0; nt < 4; ++nt) {
#pragma unroll
      for (int r = 0; r < 8; ++r) acc[nt][r] *= sc[r];
      const f16* vp =
          VT + (size_t)(h * HEADD + 16 * nt + ln) * TOTAL_KV + kb + 16 * lh;
      V16 vf;
      vf.h[0] = *(const v8h*)(vp);
      vf.h[1] = *(const v8h*)(vp + 8);
      acc[nt] = __builtin_amdgcn_wmma_f32_16x16x32_f16(
          false, pa.v, false, vf.v, (short)0, acc[nt], false, false);
    }
  }

#pragma unroll
  for (int nt = 0; nt < 4; ++nt)
#pragma unroll
    for (int r = 0; r < 8; ++r) {
      const float val = acc[nt][r] / l_i[r];
      WVh[(size_t)(q0 + 8 * lh + r) * DV + h * HEADD + 16 * nt + ln] = (f16)val;
    }
}

// ---------------------------------------------------------------------------
// Host-side orchestration (all async on `stream`, graph-capture safe).
// ---------------------------------------------------------------------------
extern "C" void kernel_launch(void* const* d_in, const int* in_sizes, int n_in,
                              void* d_out, int out_size, void* d_ws,
                              size_t ws_size, hipStream_t stream) {
  (void)in_sizes; (void)n_in; (void)out_size; (void)ws_size;
  const float* A     = (const float*)d_in[0];
  const float* B0    = (const float*)d_in[1];
  const int*   seg_q = (const int*)d_in[2];
  const int*   seg_kv= (const int*)d_in[3];
  const float* Wq    = (const float*)d_in[4];
  const float* bq    = (const float*)d_in[5];
  const float* Wk    = (const float*)d_in[6];
  const float* bk    = (const float*)d_in[7];
  const float* Wv    = (const float*)d_in[8];
  const float* bv    = (const float*)d_in[9];
  const float* Wf    = (const float*)d_in[10];
  const float* bf    = (const float*)d_in[11];
  float* out = (float*)d_out;

  f16* ws = (f16*)d_ws;
  size_t off = 0;
  auto take = [&](size_t elems) { f16* p = ws + off; off += elems; return p; };
  f16* Ah  = take((size_t)TOTAL_Q  * QIN);     // 2048x1024
  f16* Bh  = take((size_t)TOTAL_KV * KVPAD);   // 2048x1056 (zero padded)
  f16* Wqh = take((size_t)QIN   * DQK);        // 1024x1024
  f16* Wkh = take((size_t)KVPAD * DQK);        // 1056x1024 (zero padded rows)
  f16* Wvh = take((size_t)KVPAD * DV);         // 1056x1024
  f16* Wfh = take((size_t)DV * QIN);           // 1024x1024
  f16* Qh  = take((size_t)TOTAL_Q  * DQK);
  f16* Kh  = take((size_t)TOTAL_KV * DQK);
  f16* Vh  = take((size_t)TOTAL_KV * DV);
  f16* VT  = take((size_t)DV * TOTAL_KV);      // transposed V: [d][key]
  f16* WVh = take((size_t)TOTAL_Q * DV);

  // 1) Convert + zero-pad everything to f16.
  cvt_pad_f16<<<2048, 256, 0, stream>>>(A,  Ah,  TOTAL_Q,  QIN,  TOTAL_Q,  QIN);
  cvt_pad_f16<<<2048, 256, 0, stream>>>(B0, Bh,  TOTAL_KV, KVIN, TOTAL_KV, KVPAD);
  cvt_pad_f16<<<1024, 256, 0, stream>>>(Wq, Wqh, QIN,  DQK, QIN,   DQK);
  cvt_pad_f16<<<1024, 256, 0, stream>>>(Wk, Wkh, KVIN, DQK, KVPAD, DQK);
  cvt_pad_f16<<<1024, 256, 0, stream>>>(Wv, Wvh, KVIN, DV,  KVPAD, DV);
  cvt_pad_f16<<<1024, 256, 0, stream>>>(Wf, Wfh, DV,   QIN, DV,    QIN);

  // 2) Projections: Q = A*Wq+bq, K = B0*Wk+bk, V = B0*Wv+bv  (f16 outputs).
  gemm_wmma_bias<true><<<dim3(DQK / 128, TOTAL_Q / 64), 128, 0, stream>>>(
      Ah, Wqh, bq, Qh, QIN, QIN, DQK, DQK);
  gemm_wmma_bias<true><<<dim3(DQK / 128, TOTAL_KV / 64), 128, 0, stream>>>(
      Bh, Wkh, bk, Kh, KVPAD, KVPAD, DQK, DQK);
  gemm_wmma_bias<true><<<dim3(DV / 128, TOTAL_KV / 64), 128, 0, stream>>>(
      Bh, Wvh, bv, Vh, KVPAD, KVPAD, DV, DV);

  // 3) Transpose V for contiguous B-fragment loads in attention.
  transpose_f16<<<4096, 256, 0, stream>>>(Vh, VT, TOTAL_KV, DV);

  // 4) Segment-masked flash attention (one wave per 16-query tile per head).
  flash_attn_wmma<<<dim3(TOTAL_Q / 16, NHEADS), 32, 0, stream>>>(
      Qh, Kh, VT, seg_q, seg_kv, WVh);

  // 5) Final projection: out = WV * Wf + bf  (f32 output).
  gemm_wmma_bias<false><<<dim3(QIN / 128, TOTAL_Q / 64), 128, 0, stream>>>(
      WVh, Wfh, bf, out, DV, DV, QIN, QIN);
}
